// EGCVBlockS4_21569325760699
// MI455X (gfx1250) — matrix operand order repaired and checked by reading, not compiled, not run on hardware
//
#include <hip/hip_runtime.h>
#include <hip/hip_fp16.h>

typedef __attribute__((ext_vector_type(16))) _Float16 v16h;
typedef __attribute__((ext_vector_type(8)))  float    v8f;

#define LQ   4096
#define BQ   4
#define CC   96
#define DIN  192
#define KD   4
#define NS   16
#define DTRK 6
#define MLPH 384
#define TQ   (BQ*LQ)
#define XZW  (2*DIN)   // fused xp|z row width = 384

__device__ __forceinline__ float sigmoidf_(float x){ return 1.0f/(1.0f+__expf(-x)); }
__device__ __forceinline__ float siluf_(float x){ return x*sigmoidf_(x); }
__device__ __forceinline__ float softplusf_(float x){ return x>20.f ? x : log1pf(__expf(x)); }
__device__ __forceinline__ float geluf_(float x){
  float x3=x*x*x;
  return 0.5f*x*(1.f+tanhf(0.7978845608028654f*(x+0.044715f*x3)));
}
// map scan position l -> spatial index p for direction k (H=W=64, L=4096)
__device__ __forceinline__ int pos_dir(int l,int k){
  if(k==0) return l;
  if(k==1) return ((l&63)<<6)|(l>>6);
  if(k==2) return 4095-l;
  int pl=4095-l; return ((pl&63)<<6)|(pl>>6);
}

// ---- WMMA fragment helpers (layouts per CDNA5 ISA 7.12.2) ----
// A 16x32 f16: lane holds row M=lane&15; halves 0..7 -> K = 8*(lane>>4)+j,
// halves 8..15 -> K = 8*(lane>>4)+16+j  (relative to k0)
__device__ __forceinline__ v16h frag_A_lds(const _Float16* sm,int ldk,int k0){
  int lane = threadIdx.x & 31;
  const _Float16* p = sm + (size_t)(lane&15)*ldk + k0 + ((lane>>4)<<3);
  v16h a;
#pragma unroll
  for(int j=0;j<8;++j){ a[j]=p[j]; a[8+j]=p[16+j]; }
  return a;
}
__device__ __forceinline__ v16h frag_A_glb(const _Float16* base,int stride,int row0,int k0){
  int lane = threadIdx.x & 31;
  const _Float16* p = base + (size_t)(row0+(lane&15))*stride + k0 + ((lane>>4)<<3);
  v16h a;
#pragma unroll
  for(int j=0;j<8;++j){ a[j]=p[j]; a[8+j]=p[16+j]; }
  return a;
}
// B 32x16 f16 from row-major weight W[N][Kdim] (B[k][n]=W[n][k]):
// lane -> column n = n0+(lane&15); halves 0..7 -> K = k0+8*(lane>>4)+j, halves 8..15 -> K = k0+8*(lane>>4)+16+j
__device__ __forceinline__ v16h frag_B_W(const _Float16* W,int Kdim,int n0,int k0,int nmax){
  int lane = threadIdx.x & 31;
  int n = n0 + (lane&15); if(n>nmax) n=nmax;
  const _Float16* p = W + (size_t)n*Kdim + k0 + ((lane>>4)<<3);
  v16h b;
#pragma unroll
  for(int j=0;j<8;++j){ b[j]=p[j]; b[8+j]=p[16+j]; }
  return b;
}
#define WMMA_F16(a,b,c) __builtin_amdgcn_wmma_f32_16x16x32_f16(false,(a),false,(b),(short)0,(c),false,false)

// ------------------------------------------------------------------
// K0: convert all GEMM weights f32 -> f16 (once per launch, tiny)
__global__ void k_convert_weights(const float* w0,const float* w1,const float* w2,
                                  const float* w3,const float* w4,
                                  _Float16* o0,_Float16* o1,_Float16* o2,
                                  _Float16* o3,_Float16* o4){
  int i = blockIdx.x*blockDim.x + threadIdx.x;
  const int n0 = 2*DIN*CC;        // in_proj_w  (384,96)
  const int n1 = KD*38*DIN;       // x_proj_w   (4,38,192)
  const int n2 = CC*DIN;          // out_proj_w (96,192)
  const int n3 = MLPH*CC;         // fc1_w      (384,96)
  const int n4 = CC*MLPH;         // fc2_w      (96,384)
  if(i < n0){ o0[i]=(_Float16)w0[i]; return; } i -= n0;
  if(i < n1){ o1[i]=(_Float16)w1[i]; return; } i -= n1;
  if(i < n2){ o2[i]=(_Float16)w2[i]; return; } i -= n2;
  if(i < n3){ o3[i]=(_Float16)w3[i]; return; } i -= n3;
  if(i < n4){ o4[i]=(_Float16)w4[i]; }
}

// ------------------------------------------------------------------
// K1: LN1 + in_proj GEMM (96 -> 384).  16 tokens/block, 8 waves x 3 N-tiles.
// Stores into fused xz buffer (T x 384): cols 0..191 = xp, 192..383 = z.
__global__ void __launch_bounds__(256)
k_ln1_inproj(const float* __restrict__ x, const float* __restrict__ lnw,
             const float* __restrict__ lnb, const _Float16* __restrict__ W16,
             const float* __restrict__ bias, float* __restrict__ xz){
  __shared__ float    smX[16*CC];
  __shared__ _Float16 smA[16*CC];
  __shared__ float    smMu[16], smRs[16];
  const int tid = threadIdx.x;
  const int t0  = blockIdx.x*16;
  const int b   = t0 / LQ, p0 = t0 % LQ;
  // gather 16 tokens (channels-last) from NCHW input
  for(int i=tid;i<16*CC;i+=256){
    int m=i&15, c=i>>4;
    smX[m*CC+c] = x[((size_t)b*CC+c)*LQ + p0 + m];
  }
  __syncthreads();
  if(tid<16){
    float s=0.f,ss=0.f;
    for(int c=0;c<CC;++c){ float v=smX[tid*CC+c]; s+=v; ss+=v*v; }
    float mu=s/CC;
    smMu[tid]=mu; smRs[tid]=rsqrtf(ss/CC - mu*mu + 1e-5f);
  }
  __syncthreads();
  for(int i=tid;i<16*CC;i+=256){
    int m=i/CC, c=i%CC;
    smA[i]=(_Float16)((smX[i]-smMu[m])*smRs[m]*lnw[c]+lnb[c]);
  }
  __syncthreads();
  const int wv = tid>>5, lane = tid&31;
  const int row0 = (lane>>4)<<3;
  for(int nt = wv*3; nt < wv*3+3; ++nt){
    v8f acc = {};
    for(int kc=0;kc<CC;kc+=32){
      v16h a  = frag_A_lds(smA, CC, kc);
      v16h bb = frag_B_W(W16, CC, nt*16, kc, XZW-1);
      acc = WMMA_F16(a,bb,acc);
    }
    int col = nt*16 + (lane&15);
    float bcol = bias[col];
    float* pd = xz + (size_t)(t0+row0)*XZW + col;
#pragma unroll
    for(int r=0;r<8;++r) pd[(size_t)r*XZW] = acc[r] + bcol;
  }
}

// ------------------------------------------------------------------
// K2: depthwise 7x7 conv + SiLU on xp (= xz cols 0..191); emits f32 + f16
__global__ void k_dwconv_silu(const float* __restrict__ xz, const float* __restrict__ dww,
                              const float* __restrict__ dwb,
                              float* __restrict__ xc32, _Float16* __restrict__ xc16){
  int idx = blockIdx.x*blockDim.x + threadIdx.x;
  if(idx >= TQ*DIN) return;
  int d = idx % DIN, t = idx / DIN;
  int p = t % LQ, b = t / LQ;
  int h = p >> 6, w = p & 63;
  float acc = dwb[d];
  const float* wk = dww + d*49;
  const float* xpb = xz + (size_t)b*LQ*XZW + d;
#pragma unroll
  for(int ky=0;ky<7;++ky){
    int iy = h + ky - 3; if(iy<0||iy>=64) continue;
#pragma unroll
    for(int kx=0;kx<7;++kx){
      int ix = w + kx - 3; if(ix<0||ix>=64) continue;
      acc = fmaf(xpb[(size_t)((iy<<6)+ix)*XZW], wk[ky*7+kx], acc);
    }
  }
  float s = siluf_(acc);
  xc32[idx]=s; xc16[idx]=(_Float16)s;
}

// ------------------------------------------------------------------
// K3: x_proj GEMM (192 -> 38, padded to 48) per direction; gather tokens by scan order.
__global__ void __launch_bounds__(96)
k_xproj(const _Float16* __restrict__ xc16, const _Float16* __restrict__ Wx,
        float* __restrict__ xdbl){
  __shared__ _Float16 smA[16*DIN];
  __shared__ int smPos[16];
  const int tid  = threadIdx.x;
  const int bk   = blockIdx.x >> 8;     // 256 tiles per (b,k)
  const int tile = blockIdx.x & 255;
  const int b = bk >> 2, k = bk & 3;
  if(tid<16) smPos[tid] = pos_dir(tile*16+tid, k);
  __syncthreads();
  for(int i=tid;i<16*DIN;i+=96){
    int m = i / DIN, d = i % DIN;
    smA[i] = xc16[((size_t)(b*LQ) + smPos[m])*DIN + d];
  }
  __syncthreads();
  const int wv = tid >> 5, lane = tid & 31;
  v8f acc = {};
  const _Float16* W = Wx + (size_t)k*38*DIN;
  for(int kc=0;kc<DIN;kc+=32){
    v16h a  = frag_A_lds(smA, DIN, kc);
    v16h bb = frag_B_W(W, DIN, wv*16, kc, 37);
    acc = WMMA_F16(a,bb,acc);
  }
  int col = wv*16 + (lane&15);
  if(col<38){
    const int row0 = (lane>>4)<<3;
    float* pd = xdbl + ((size_t)bk*LQ + tile*16 + row0)*48 + col;
#pragma unroll
    for(int r=0;r<8;++r) pd[r*48] = acc[r];
  }
}

// ------------------------------------------------------------------
// K4: selective scan. 1 block per (b,k); 1 thread per channel d (N=16 state in regs).
// dt projection (rank 6) folded in; merged y accumulated with fp32 global atomics.
__global__ void __launch_bounds__(192)
k_scan(const float* __restrict__ xdbl, const float* __restrict__ xc32,
       const float* __restrict__ A_log, const float* __restrict__ dtw_g,
       const float* __restrict__ dtb_g, float* __restrict__ ym){
  __shared__ float smRow[40];
  const int d  = threadIdx.x;
  const int bk = blockIdx.x;
  const int b  = bk >> 2, k = bk & 3;
  float Anr[NS];
#pragma unroll
  for(int n=0;n<NS;++n) Anr[n] = -__expf(A_log[((size_t)(k*DIN)+d)*NS+n]);
  float dtw[DTRK];
#pragma unroll
  for(int r=0;r<DTRK;++r) dtw[r] = dtw_g[((size_t)(k*DIN)+d)*DTRK+r];
  const float dtb = dtb_g[k*DIN+d];
  float h[NS];
#pragma unroll
  for(int n=0;n<NS;++n) h[n]=0.f;
  const float* rowbase = xdbl + (size_t)bk*LQ*48;
  for(int l=0;l<LQ;++l){
    __syncthreads();
    if(d<38) smRow[d] = rowbase[(size_t)l*48 + d];
    __syncthreads();
    int p = pos_dir(l,k);
    float xt = xc32[((size_t)(b*LQ)+p)*DIN + d];
    float dt = dtb;
#pragma unroll
    for(int r=0;r<DTRK;++r) dt = fmaf(smRow[r], dtw[r], dt);
    dt = softplusf_(dt);
    float s = dt*xt;
    float y = 0.f;
#pragma unroll
    for(int n=0;n<NS;++n){
      float dA = __expf(dt*Anr[n]);
      h[n] = fmaf(dA, h[n], s*smRow[6+n]);
      y = fmaf(h[n], smRow[22+n], y);
    }
    unsafeAtomicAdd(&ym[((size_t)(b*LQ)+p)*DIN + d], y);  // global_atomic_add_f32
  }
}

// ------------------------------------------------------------------
// K5: y = out_LN(ym + (sum_k Dskip)*xc) * silu(z) -> f16.  1 wave per token.
__global__ void __launch_bounds__(256)
k_gate(const float* __restrict__ ym, const float* __restrict__ xc32,
       const float* __restrict__ xz, const float* __restrict__ Dskip,
       const float* __restrict__ olnw, const float* __restrict__ olnb,
       _Float16* __restrict__ yg){
  const int lane = threadIdx.x & 31;
  const int t = blockIdx.x*8 + (threadIdx.x >> 5);
  const float* zrow = xz + (size_t)t*XZW + DIN;   // z half of fused buffer
  float v[6], s=0.f, ss=0.f;
#pragma unroll
  for(int i=0;i<6;++i){
    int d = lane + i*32;
    float ds = Dskip[d] + Dskip[DIN+d] + Dskip[2*DIN+d] + Dskip[3*DIN+d];
    float val = ym[(size_t)t*DIN+d] + ds*xc32[(size_t)t*DIN+d];
    v[i]=val; s+=val; ss+=val*val;
  }
#pragma unroll
  for(int o=16;o>0;o>>=1){ s += __shfl_xor(s,o,32); ss += __shfl_xor(ss,o,32); }
  float mu = s/DIN;
  float rs = rsqrtf(ss/DIN - mu*mu + 1e-5f);
#pragma unroll
  for(int i=0;i<6;++i){
    int d = lane + i*32;
    float g = (v[i]-mu)*rs*olnw[d] + olnb[d];
    g *= siluf_(zrow[d]);
    yg[(size_t)t*DIN+d] = (_Float16)g;
  }
}

// ------------------------------------------------------------------
// K6: out_proj GEMM (192 -> 96) + bias + residual(xl) -> x2 (B,L,96)
__global__ void __launch_bounds__(192)
k_outproj(const _Float16* __restrict__ yg, const _Float16* __restrict__ W16,
          const float* __restrict__ bias, const float* __restrict__ xin,
          float* __restrict__ x2){
  const int t0 = blockIdx.x*16;
  const int b = t0 / LQ, p0 = t0 % LQ;
  const int wv = threadIdx.x >> 5, lane = threadIdx.x & 31;
  v8f acc = {};
  for(int kc=0;kc<DIN;kc+=32){
    v16h a  = frag_A_glb(yg, DIN, t0, kc);
    v16h bb = frag_B_W(W16, DIN, wv*16, kc, CC-1);
    acc = WMMA_F16(a,bb,acc);
  }
  int col = wv*16 + (lane&15);
  float bcol = bias[col];
  const int row0 = (lane>>4)<<3;
  const float* pres = xin + ((size_t)b*CC+col)*LQ + p0 + row0;   // +1 float per row
  float* pd = x2 + (size_t)(t0+row0)*CC + col;                    // +CC per row
#pragma unroll
  for(int r=0;r<8;++r) pd[(size_t)r*CC] = acc[r] + bcol + pres[r];
}

// ------------------------------------------------------------------
// K7: LN2 + fc1 GEMM (96 -> 384) + GELU -> f16
__global__ void __launch_bounds__(256)
k_ln2_fc1(const float* __restrict__ x2, const float* __restrict__ lnw,
          const float* __restrict__ lnb, const _Float16* __restrict__ W16,
          const float* __restrict__ bias, _Float16* __restrict__ m16){
  __shared__ float    smX[16*CC];
  __shared__ _Float16 smA[16*CC];
  __shared__ float    smMu[16], smRs[16];
  const int tid = threadIdx.x;
  const int t0  = blockIdx.x*16;
  for(int i=tid;i<16*CC;i+=256) smX[i] = x2[(size_t)t0*CC + i];
  __syncthreads();
  if(tid<16){
    float s=0.f,ss=0.f;
    for(int c=0;c<CC;++c){ float v=smX[tid*CC+c]; s+=v; ss+=v*v; }
    float mu=s/CC;
    smMu[tid]=mu; smRs[tid]=rsqrtf(ss/CC - mu*mu + 1e-5f);
  }
  __syncthreads();
  for(int i=tid;i<16*CC;i+=256){
    int m=i/CC, c=i%CC;
    smA[i]=(_Float16)((smX[i]-smMu[m])*smRs[m]*lnw[c]+lnb[c]);
  }
  __syncthreads();
  const int wv = tid>>5, lane = tid&31;
  const int row0 = (lane>>4)<<3;
  for(int nt = wv*3; nt < wv*3+3; ++nt){
    v8f acc = {};
    for(int kc=0;kc<CC;kc+=32){
      v16h a  = frag_A_lds(smA, CC, kc);
      v16h bb = frag_B_W(W16, CC, nt*16, kc, MLPH-1);
      acc = WMMA_F16(a,bb,acc);
    }
    int col = nt*16 + (lane&15);
    float bcol = bias[col];
    _Float16* pd = m16 + (size_t)(t0+row0)*MLPH + col;
#pragma unroll
    for(int r=0;r<8;++r) pd[(size_t)r*MLPH] = (_Float16)geluf_(acc[r] + bcol);
  }
}

// ------------------------------------------------------------------
// K8: fc2 GEMM (384 -> 96) + bias + residual(x2), scatter to NCHW output
__global__ void __launch_bounds__(192)
k_fc2_out(const _Float16* __restrict__ m16, const _Float16* __restrict__ W16,
          const float* __restrict__ bias, const float* __restrict__ x2,
          float* __restrict__ out){
  const int t0 = blockIdx.x*16;
  const int b = t0 / LQ, p0 = t0 % LQ;
  const int wv = threadIdx.x >> 5, lane = threadIdx.x & 31;
  v8f acc = {};
  for(int kc=0;kc<MLPH;kc+=32){
    v16h a  = frag_A_glb(m16, MLPH, t0, kc);
    v16h bb = frag_B_W(W16, MLPH, wv*16, kc, CC-1);
    acc = WMMA_F16(a,bb,acc);
  }
  int col = wv*16 + (lane&15);
  float bcol = bias[col];
  const int row0 = (lane>>4)<<3;
  const float* pres = x2 + (size_t)(t0+row0)*CC + col;            // +CC per row
  float* pd = out + ((size_t)b*CC+col)*LQ + p0 + row0;            // +1 per row
#pragma unroll
  for(int r=0;r<8;++r) pd[r] = acc[r] + bcol + pres[(size_t)r*CC];
}

// ------------------------------------------------------------------
extern "C" void kernel_launch(void* const* d_in, const int* in_sizes, int n_in,
                              void* d_out, int out_size, void* d_ws, size_t ws_size,
                              hipStream_t stream) {
  (void)in_sizes; (void)n_in; (void)out_size; (void)ws_size;
  const float* x         = (const float*)d_in[0];
  const float* ln1_w     = (const float*)d_in[1];
  const float* ln1_b     = (const float*)d_in[2];
  const float* in_proj_w = (const float*)d_in[3];
  const float* in_proj_b = (const float*)d_in[4];
  const float* dw_w      = (const float*)d_in[5];
  const float* dw_b      = (const float*)d_in[6];
  const float* x_proj_w  = (const float*)d_in[7];
  const float* dt_proj_w = (const float*)d_in[8];
  const float* dt_proj_b = (const float*)d_in[9];
  const float* A_log     = (const float*)d_in[10];
  const float* Dskip     = (const float*)d_in[11];
  const float* out_ln_w  = (const float*)d_in[12];
  const float* out_ln_b  = (const float*)d_in[13];
  const float* out_proj_w= (const float*)d_in[14];
  const float* out_proj_b= (const float*)d_in[15];
  const float* ln2_w     = (const float*)d_in[16];
  const float* ln2_b     = (const float*)d_in[17];
  const float* fc1_w     = (const float*)d_in[18];
  const float* fc1_b     = (const float*)d_in[19];
  const float* fc2_w     = (const float*)d_in[20];
  const float* fc2_b     = (const float*)d_in[21];
  float* out = (float*)d_out;

  char* ws = (char*)d_ws; size_t off = 0;
  auto alloc = [&](size_t bytes)->char*{
    char* p = ws + off; off = (off + bytes + 255) & ~(size_t)255; return p;
  };
  _Float16* w16_in  = (_Float16*)alloc((size_t)2*DIN*CC*2);
  _Float16* w16_x   = (_Float16*)alloc((size_t)KD*38*DIN*2);
  _Float16* w16_out = (_Float16*)alloc((size_t)CC*DIN*2);
  _Float16* w16_fc1 = (_Float16*)alloc((size_t)MLPH*CC*2);
  _Float16* w16_fc2 = (_Float16*)alloc((size_t)CC*MLPH*2);
  float*    xzbuf   = (float*)   alloc((size_t)TQ*XZW*4);   // fused xp|z
  float*    xc32    = (float*)   alloc((size_t)TQ*DIN*4);
  _Float16* xc16    = (_Float16*)alloc((size_t)TQ*DIN*2);
  float*    xdbl    = (float*)   alloc((size_t)BQ*KD*LQ*48*4);
  float*    ym      = (float*)   alloc((size_t)TQ*DIN*4);
  _Float16* yg      = (_Float16*)alloc((size_t)TQ*DIN*2);
  float*    x2      = (float*)   alloc((size_t)TQ*CC*4);
  _Float16* m16     = (_Float16*)alloc((size_t)TQ*MLPH*2);

  // weight conversion + zero the atomic accumulator
  {
    int total = 2*DIN*CC + KD*38*DIN + CC*DIN + MLPH*CC + CC*MLPH;
    k_convert_weights<<<(total+255)/256, 256, 0, stream>>>(
        in_proj_w, x_proj_w, out_proj_w, fc1_w, fc2_w,
        w16_in, w16_x, w16_out, w16_fc1, w16_fc2);
    hipMemsetAsync(ym, 0, (size_t)TQ*DIN*4, stream);
  }
  k_ln1_inproj<<<TQ/16, 256, 0, stream>>>(x, ln1_w, ln1_b, w16_in, in_proj_b, xzbuf);
  k_dwconv_silu<<<(TQ*DIN+255)/256, 256, 0, stream>>>(xzbuf, dw_w, dw_b, xc32, xc16);
  k_xproj<<<BQ*KD*(LQ/16), 96, 0, stream>>>(xc16, w16_x, xdbl);
  k_scan<<<BQ*KD, DIN, 0, stream>>>(xdbl, xc32, A_log, dt_proj_w, dt_proj_b, ym);
  k_gate<<<TQ/8, 256, 0, stream>>>(ym, xc32, xzbuf, Dskip, out_ln_w, out_ln_b, yg);
  k_outproj<<<TQ/16, 192, 0, stream>>>(yg, w16_out, out_proj_b, x, x2);
  k_ln2_fc1<<<TQ/16, 256, 0, stream>>>(x2, ln2_w, ln2_b, w16_fc1, fc1_b, m16);
  k_fc2_out<<<TQ/16, 192, 0, stream>>>(m16, w16_fc2, fc2_b, x2, out);
}
